// GRUAutoencoder_24670292148755
// MI455X (gfx1250) — matrix operand order repaired and checked by reading, not compiled, not run on hardware
//
#include <hip/hip_runtime.h>
#include <hip/hip_bf16.h>
#include <math.h>

// GRU autoencoder, persistent-recurrence design for MI455X (gfx1250, wave32).
// B=512,T=1024,D=5,H=128,L=16. One workgroup (8 wave32) owns 16 batch rows and
// runs the full time loop; W_hh lives in registers as f16 WMMA B-fragments.
// Decoder streams h_t to scratch; output projection is a separate WMMA GEMM.

#define BB 512
#define TT 1024
#define DD 5
#define HH 128
#define LL 16
#define G3 384          // 3*H
#define BT 16           // batch rows per workgroup (WMMA M)
#define NTPW 3          // N tiles (of 16) per wave: 8 waves * 3 = 24 = 384/16
#define H16S 136        // padded f16 h row stride (breaks LDS bank aliasing)
#define HPS 388         // padded f32 hp/xp row stride

typedef __attribute__((ext_vector_type(16))) _Float16 v16h;
typedef __attribute__((ext_vector_type(8)))  float    v8f;

__device__ __forceinline__ float sigm(float x) { return 1.0f / (1.0f + __expf(-x)); }

// Build persistent f16 B-operand fragments of W_hh^T (W_hh is [384][128] row major;
// B[k][n] = W_hh[n][k]).  Layout per CDNA5 ISA: lane n=lane&15 holds column n,
// lanes>=16 hold K+8; VGPR v<4 holds K pair (2v, 2v+1), v>=4 holds (16+2(v-4), +1).
__device__ __forceinline__ void load_bfrags(const float* __restrict__ W_hh,
                                            int wave, int lane, v16h (&Bf)[NTPW][4]) {
  const int col  = lane & 15;
  const int hsel = lane >> 4;
#pragma unroll
  for (int j = 0; j < NTPW; ++j) {
    const int n = (wave * NTPW + j) * 16 + col;
#pragma unroll
    for (int kc = 0; kc < 4; ++kc) {
#pragma unroll
      for (int v = 0; v < 8; ++v) {
        const int k = kc * 32 + ((v < 4) ? 2 * v : 16 + 2 * (v - 4)) + hsel * 8;
        Bf[j][kc][2 * v]     = (_Float16)W_hh[n * HH + k];
        Bf[j][kc][2 * v + 1] = (_Float16)W_hh[n * HH + k + 1];
      }
    }
  }
}

// ---------------------------------------------------------------------------
// Encoder: per step hp = h @ W_hh^T (WMMA), gates in f32, h kept as f16 in LDS.
// ---------------------------------------------------------------------------
__global__ __launch_bounds__(256) void gru_encoder_kernel(
    const float* __restrict__ x,      // [B,T,D]
    const float* __restrict__ W_ih,   // [384,5]
    const float* __restrict__ W_hh,   // [384,128]
    const float* __restrict__ b_ih,   // [384]
    const float* __restrict__ b_hh,   // [384]
    float* __restrict__ g_hn)         // [B,128]
{
  __shared__ _Float16 h16[BT][H16S];
  __shared__ float hp[BT][HPS];
  __shared__ float wih[G3 * DD];
  __shared__ float bih[G3];
  __shared__ float bhh[G3];
  __shared__ float xt[BT * DD];

  const int tid  = threadIdx.x;
  const int wave = tid >> 5;
  const int lane = tid & 31;
  const int col  = lane & 15;
  const int hsel = lane >> 4;
  const int b0   = blockIdx.x * BT;

  for (int i = tid; i < G3 * DD; i += 256) wih[i] = W_ih[i];
  for (int i = tid; i < G3; i += 256) { bih[i] = b_ih[i]; bhh[i] = b_hh[i]; }
  for (int i = tid; i < BT * H16S; i += 256) (&h16[0][0])[i] = (_Float16)0.0f;

  v16h Bf[NTPW][4];
  load_bfrags(W_hh, wave, lane, Bf);
  __syncthreads();

  for (int t = 0; t < TT; ++t) {
    // Stage x[:,t,:] for this batch tile (80 values), prefetch next step's row.
    if (tid < BT * DD) {
      const size_t gofs = (size_t)(b0 + tid / DD) * TT * DD + (size_t)t * DD + tid % DD;
      xt[tid] = x[gofs];
      if (t + 1 < TT) __builtin_prefetch(&x[gofs + DD], 0, 1);  // global_prefetch_b8
    }
    __syncthreads();

    // hp[16,384] = h[16,128] @ W_hh^T   (WMMA f16 -> f32)
    v8f acc[NTPW] = {};
#pragma unroll
    for (int kc = 0; kc < 4; ++kc) {
      v16h a;
#pragma unroll
      for (int v = 0; v < 8; ++v) {
        const int k = kc * 32 + ((v < 4) ? 2 * v : 16 + 2 * (v - 4)) + hsel * 8;
        a[2 * v]     = h16[col][k];
        a[2 * v + 1] = h16[col][k + 1];
      }
#pragma unroll
      for (int j = 0; j < NTPW; ++j)
        acc[j] = __builtin_amdgcn_wmma_f32_16x16x32_f16(
            false, a, false, Bf[j][kc], (short)0, acc[j], false, false);
    }
#pragma unroll
    for (int j = 0; j < NTPW; ++j) {
      const int n = (wave * NTPW + j) * 16 + col;
#pragma unroll
      for (int v = 0; v < 8; ++v)
        hp[v + 8 * hsel][n] = acc[j][v];
    }
    __syncthreads();

    // Gates: 2048 elements over 256 threads.
#pragma unroll
    for (int e = 0; e < (BT * HH) / 256; ++e) {
      const int idx = tid + e * 256;
      const int m = idx >> 7, i = idx & 127;
      float xp0 = bih[i], xp1 = bih[HH + i], xp2 = bih[2 * HH + i];
#pragma unroll
      for (int d = 0; d < DD; ++d) {
        const float xv = xt[m * DD + d];
        xp0 += xv * wih[i * DD + d];
        xp1 += xv * wih[(HH + i) * DD + d];
        xp2 += xv * wih[(2 * HH + i) * DD + d];
      }
      const float r  = sigm(xp0 + hp[m][i] + bhh[i]);
      const float zg = sigm(xp1 + hp[m][HH + i] + bhh[HH + i]);
      const float nn = tanhf(xp2 + r * (hp[m][2 * HH + i] + bhh[2 * HH + i]));
      const float ho = (float)h16[m][i];
      h16[m][i] = (_Float16)((1.0f - zg) * nn + zg * ho);
    }
    __syncthreads();
  }

  for (int idx = tid; idx < BT * HH; idx += 256)
    g_hn[(size_t)(b0 + (idx >> 7)) * HH + (idx & 127)] =
        (float)h16[idx >> 7][idx & 127];
}

// ---------------------------------------------------------------------------
// Decoder: prologue computes z, h_dec_in, constant xp_dec (b_hh r/z folded in);
// time loop mirrors the encoder and streams h_t (f16) to scratch for the
// output-projection GEMM.  Only 2 barriers per step.
// ---------------------------------------------------------------------------
__global__ __launch_bounds__(256) void gru_decoder_kernel(
    const float* __restrict__ g_hn,   // [B,128] encoder final hidden
    const float* __restrict__ efc_W,  // [16,128]
    const float* __restrict__ efc_b,  // [16]
    const float* __restrict__ dfc_W,  // [128,16]
    const float* __restrict__ dfc_b,  // [128]
    const float* __restrict__ W_ih,   // [384,128]
    const float* __restrict__ b_ih,   // [384]
    const float* __restrict__ W_hh,   // [384,128]
    const float* __restrict__ b_hh,   // [384]
    _Float16* __restrict__ g_hs,      // [B*T,128] f16 hidden history (scratch)
    float* __restrict__ z_out)        // [B,16]
{
  __shared__ _Float16 h16[BT][H16S];
  __shared__ float hp[BT][HPS];
  __shared__ float xp[BT][HPS];      // constant input projection (b_ih + r/z b_hh)
  __shared__ float bhh_n[HH];        // n-gate b_hh stays separate (multiplied by r)
  __shared__ float zl[BT][LL];

  const int tid  = threadIdx.x;
  const int wave = tid >> 5;
  const int lane = tid & 31;
  const int col  = lane & 15;
  const int hsel = lane >> 4;
  const int b0   = blockIdx.x * BT;

  for (int i = tid; i < HH; i += 256) bhh_n[i] = b_hh[2 * HH + i];
  for (int i = tid; i < BT * H16S; i += 256) (&h16[0][0])[i] = (_Float16)0.0f;
  // Stage encoder h_n into hp[:, 0:128].
  for (int idx = tid; idx < BT * HH; idx += 256)
    hp[idx >> 7][idx & 127] = g_hn[(size_t)(b0 + (idx >> 7)) * HH + (idx & 127)];

  v16h Bf[NTPW][4];
  load_bfrags(W_hh, wave, lane, Bf);
  __syncthreads();

  // z = h_n @ efc_W^T + b  -> zl and output tail.
  {
    const int m = tid >> 4, l = tid & 15;
    float a = efc_b[l];
    for (int k = 0; k < HH; ++k) a += hp[m][k] * efc_W[l * HH + k];
    zl[m][l] = a;
    z_out[(size_t)(b0 + m) * LL + l] = a;
  }
  __syncthreads();
  // h_dec_in = z @ dfc_W^T + b  -> reuse hp[:, 0:128] (h_n dead now).
  for (int idx = tid; idx < BT * HH; idx += 256) {
    const int m = idx >> 7, j = idx & 127;
    float a = dfc_b[j];
#pragma unroll
    for (int l = 0; l < LL; ++l) a += zl[m][l] * dfc_W[j * LL + l];
    hp[m][j] = a;
  }
  __syncthreads();
  // xp_dec = h_dec_in @ W_ih^T + b_ih (+ b_hh folded for r/z gates).
  for (int idx = tid; idx < BT * G3; idx += 256) {
    const int m = idx / G3, g = idx % G3;
    float a = b_ih[g] + ((g < 2 * HH) ? b_hh[g] : 0.0f);
    for (int j = 0; j < HH; ++j) a += hp[m][j] * W_ih[g * HH + j];
    xp[m][g] = a;
  }
  __syncthreads();

  for (int t = 0; t < TT; ++t) {
    v8f acc[NTPW] = {};
#pragma unroll
    for (int kc = 0; kc < 4; ++kc) {
      v16h a;
#pragma unroll
      for (int v = 0; v < 8; ++v) {
        const int k = kc * 32 + ((v < 4) ? 2 * v : 16 + 2 * (v - 4)) + hsel * 8;
        a[2 * v]     = h16[col][k];
        a[2 * v + 1] = h16[col][k + 1];
      }
#pragma unroll
      for (int j = 0; j < NTPW; ++j)
        acc[j] = __builtin_amdgcn_wmma_f32_16x16x32_f16(
            false, a, false, Bf[j][kc], (short)0, acc[j], false, false);
    }
#pragma unroll
    for (int j = 0; j < NTPW; ++j) {
      const int n = (wave * NTPW + j) * 16 + col;
#pragma unroll
      for (int v = 0; v < 8; ++v)
        hp[v + 8 * hsel][n] = acc[j][v];
    }
    __syncthreads();

#pragma unroll
    for (int e = 0; e < (BT * HH) / 256; ++e) {
      const int idx = tid + e * 256;
      const int m = idx >> 7, i = idx & 127;
      const float r  = sigm(xp[m][i] + hp[m][i]);
      const float zg = sigm(xp[m][HH + i] + hp[m][HH + i]);
      const float nn = tanhf(xp[m][2 * HH + i] + r * (hp[m][2 * HH + i] + bhh_n[i]));
      const float ho = (float)h16[m][i];
      const _Float16 hv = (_Float16)((1.0f - zg) * nn + zg * ho);
      h16[m][i] = hv;
      // Stream hidden state to scratch (coalesced 2B stores; off critical path).
      g_hs[((size_t)(b0 + m) * TT + t) * HH + i] = hv;
    }
    __syncthreads();
  }
}

// ---------------------------------------------------------------------------
// Output projection: recon[B*T,5] = hs[B*T,128] @ out_W^T + out_b, via WMMA
// (N padded 5->16).  Each wave owns one M=16 tile; 8 tiles per workgroup.
// ---------------------------------------------------------------------------
__global__ __launch_bounds__(256) void out_proj_kernel(
    const _Float16* __restrict__ g_hs,  // [B*T,128]
    const float* __restrict__ out_W,    // [5,128]
    const float* __restrict__ out_b,    // [5]
    float* __restrict__ recon)          // [B*T,5]
{
  const int tid  = threadIdx.x;
  const int wave = tid >> 5;
  const int lane = tid & 31;
  const int col  = lane & 15;
  const int hsel = lane >> 4;
  const size_t r0 = ((size_t)blockIdx.x * 8 + wave) * BT;  // first output row

  v8f acc = {};
#pragma unroll
  for (int kc = 0; kc < 4; ++kc) {
    v16h a, b;
#pragma unroll
    for (int v = 0; v < 8; ++v) {
      const int ka = kc * 32 + ((v < 4) ? 2 * v : 16 + 2 * (v - 4)) + hsel * 8;
      a[2 * v]     = g_hs[(r0 + col) * HH + ka];
      a[2 * v + 1] = g_hs[(r0 + col) * HH + ka + 1];
      const int kb = kc * 32 + ((v < 4) ? 2 * v : 16 + 2 * (v - 4)) + hsel * 8;
      b[2 * v]     = (col < DD) ? (_Float16)out_W[col * HH + kb]     : (_Float16)0.0f;
      b[2 * v + 1] = (col < DD) ? (_Float16)out_W[col * HH + kb + 1] : (_Float16)0.0f;
    }
    acc = __builtin_amdgcn_wmma_f32_16x16x32_f16(false, a, false, b, (short)0,
                                                 acc, false, false);
  }
  if (col < DD) {
    const float bias = out_b[col];
#pragma unroll
    for (int v = 0; v < 8; ++v) {
      const int m = v + 8 * hsel;
      recon[(r0 + m) * DD + col] = acc[v] + bias;
    }
  }
}

extern "C" void kernel_launch(void* const* d_in, const int* in_sizes, int n_in,
                              void* d_out, int out_size, void* d_ws, size_t ws_size,
                              hipStream_t stream) {
  const float* x        = (const float*)d_in[0];
  const float* enc_W_ih = (const float*)d_in[1];
  const float* enc_W_hh = (const float*)d_in[2];
  const float* enc_b_ih = (const float*)d_in[3];
  const float* enc_b_hh = (const float*)d_in[4];
  const float* enc_fc_W = (const float*)d_in[5];
  const float* enc_fc_b = (const float*)d_in[6];
  const float* dec_fc_W = (const float*)d_in[7];
  const float* dec_fc_b = (const float*)d_in[8];
  const float* dec_W_ih = (const float*)d_in[9];
  const float* dec_b_ih = (const float*)d_in[10];
  const float* dec_W_hh = (const float*)d_in[11];
  const float* dec_b_hh = (const float*)d_in[12];
  const float* out_W    = (const float*)d_in[13];
  const float* out_b    = (const float*)d_in[14];

  float*     g_hn  = (float*)d_ws;                              // [B,128]
  _Float16*  g_hs  = (_Float16*)((char*)d_ws +
                                 (size_t)BB * HH * sizeof(float));  // [B*T,128]
  float* recon = (float*)d_out;                                 // [B,T,D]
  float* z_out = (float*)d_out + (size_t)BB * TT * DD;          // [B,16]

  dim3 block(256);
  gru_encoder_kernel<<<dim3(BB / BT), block, 0, stream>>>(
      x, enc_W_ih, enc_W_hh, enc_b_ih, enc_b_hh, g_hn);
  gru_decoder_kernel<<<dim3(BB / BT), block, 0, stream>>>(
      g_hn, enc_fc_W, enc_fc_b, dec_fc_W, dec_fc_b,
      dec_W_ih, dec_b_ih, dec_W_hh, dec_b_hh, g_hs, z_out);
  out_proj_kernel<<<dim3((BB * TT) / (BT * 8)), block, 0, stream>>>(
      g_hs, out_W, out_b, recon);
}
